// GAE_20864951123973
// MI455X (gfx1250) — matrix operand order repaired and checked by reading, not compiled, not run on hardware
//
#include <hip/hip_runtime.h>
#include <hip/hip_bf16.h>

// ---------------------------------------------------------------------------
// GCN-GAE pipeline for MI455X (gfx1250, wave32, WMMA):
//   agg0 = scatter_add(feat[src] -> dst)           (fp32 atomics, L2 resident)
//   h    = relu(agg0 @ W0 + b0)                    (bf16 WMMA, f32 acc)
//   agg1 = scatter_add(h[src] -> dst)              (fp32 atomics)
//   z    = (agg1 @ W1 + b1) * drop_mask            (bf16 WMMA, f32 acc)
//   adj  = z @ z.T                                 (bf16 WMMA, f32 acc, 1GB out)
// Final GEMM is HBM-store bound (~1GB / 23.3 TB/s ~ 46us); z (1MB bf16) is
// L2-resident, so adj stores use non-temporal hints to avoid evicting it.
// ---------------------------------------------------------------------------

typedef __attribute__((ext_vector_type(16))) __bf16 v16bf;
typedef __attribute__((ext_vector_type(8)))  float  v8f;

union Frag32B { v16bf v; uint4 u[2]; };

// ---------------- utility kernels ----------------

__global__ void zero_f32(float* __restrict__ p, size_t n) {
    size_t i = (size_t)blockIdx.x * blockDim.x + threadIdx.x;
    size_t stride = (size_t)gridDim.x * blockDim.x;
    for (; i < n; i += stride) p[i] = 0.0f;
}

__global__ void cvt_f32_to_bf16(const float* __restrict__ x, __bf16* __restrict__ y, size_t n) {
    size_t i = (size_t)blockIdx.x * blockDim.x + threadIdx.x;
    size_t stride = (size_t)gridDim.x * blockDim.x;
    for (; i < n; i += stride) y[i] = (__bf16)x[i];
}

// W: K x N row-major (fp32)  ->  Wt: N x K row-major (bf16)
__global__ void transpose_to_bf16(const float* __restrict__ W, __bf16* __restrict__ Wt,
                                  int K, int N) {
    int idx = blockIdx.x * blockDim.x + threadIdx.x;
    if (idx < K * N) {
        int k = idx / N, n = idx % N;
        Wt[(size_t)n * K + k] = (__bf16)W[(size_t)k * N + n];
    }
}

// One block per edge; D threads accumulate one feature row.
__global__ void scatter_add_edges(const float* __restrict__ h,
                                  const int* __restrict__ src,
                                  const int* __restrict__ dst,
                                  float* __restrict__ agg, int D) {
    int e = blockIdx.x;
    int s = src[e], d = dst[e];
    const float* __restrict__ row = h + (size_t)s * D;
    float* __restrict__ out = agg + (size_t)d * D;
    for (int i = threadIdx.x; i < D; i += blockDim.x)
        atomicAdd(&out[i], row[i]);
}

// ---------------- generic bf16 WMMA GEMM:  C = act(A @ Bt^T + bias) ----------------
// A  : M x KC row-major bf16
// Bt : N x KC row-major bf16 (B stored transposed -> contiguous-K b128 loads)
// One wave computes a 16 x (16*NT) strip; A-fragment reused across NT col tiles.
// KC is compile-time: K-loop fully unrolled, offsets folded into load immediates.
// NTST: use non-temporal stores for outF (streaming 1GB output, written once).

template<int NT, bool RELU, int KC, bool NTST>
__global__ __launch_bounds__(256) void wmma_gemm_bt(
    const __bf16* __restrict__ A,
    const __bf16* __restrict__ Bt,
    const float*  __restrict__ bias,   // len N, may be null
    const float*  __restrict__ mask,   // M x N elementwise, may be null
    float*  __restrict__ outF,         // may be null
    __bf16* __restrict__ outB,         // may be null
    int M, int N)
{
    const int lane = threadIdx.x & 31;
    const int wave = threadIdx.x >> 5;
    const int half = lane >> 4;       // 0: lanes 0-15, 1: lanes 16-31
    const int l16  = lane & 15;

    const int gw     = blockIdx.x * (blockDim.x >> 5) + wave;
    const int tilesN = N / (16 * NT);
    const int tm     = gw / tilesN;
    const int tn0    = (gw % tilesN) * NT;
    if (tm * 16 >= M) return;

    v8f zero = {};
    v8f acc[NT];
#pragma unroll
    for (int t = 0; t < NT; ++t) acc[t] = zero;

    const int mrow = tm * 16 + l16;

#pragma unroll
    for (int k0 = 0; k0 < KC; k0 += 32) {
        // A fragment 16x32 bf16 (ISA 7.12.2 layout):
        //   lane half=0: VGPR0-3 = K[0..8), VGPR4-7 = K[16..24)
        //   lane half=1: VGPR0-3 = K[8..16), VGPR4-7 = K[24..32)
        Frag32B a;
        const __bf16* ap = A + (size_t)mrow * KC + k0;
        a.u[0] = *(const uint4*)(ap + half * 8);
        a.u[1] = *(const uint4*)(ap + 16 + half * 8);

#pragma unroll
        for (int t = 0; t < NT; ++t) {
            // B fragment 32x16: column n = l16, K = [8*half, 8*half+8) then [16+8*half, 16+8*half+8)
            const int ncol = (tn0 + t) * 16 + l16;
            Frag32B b;
            const __bf16* bp = Bt + (size_t)ncol * KC + k0 + half * 8;
            b.u[0] = *(const uint4*)(bp);
            b.u[1] = *(const uint4*)(bp + 16);

            acc[t] = __builtin_amdgcn_wmma_f32_16x16x32_bf16(
                /*neg_a=*/false, a.v, /*neg_b=*/false, b.v,
                /*c_mod=*/(short)0, acc[t], /*reuse_a=*/false, /*reuse_b=*/false);
        }
    }

    // Epilogue: C layout -> VGPR r holds row 8*half + r, col = l16
#pragma unroll
    for (int t = 0; t < NT; ++t) {
        const int nglob = (tn0 + t) * 16 + l16;
        const float bv = bias ? bias[nglob] : 0.0f;
#pragma unroll
        for (int r = 0; r < 8; ++r) {
            const int m = tm * 16 + half * 8 + r;
            float v = acc[t][r] + bv;
            if (RELU) v = v > 0.0f ? v : 0.0f;
            if (mask) v *= mask[(size_t)m * N + nglob];
            const size_t oi = (size_t)m * N + nglob;
            if (outF) {
                if (NTST) __builtin_nontemporal_store(v, &outF[oi]);
                else      outF[oi] = v;
            }
            if (outB) outB[oi] = (__bf16)v;
        }
    }
}

// ---------------- host-side orchestration ----------------

extern "C" void kernel_launch(void* const* d_in, const int* in_sizes, int n_in,
                              void* d_out, int out_size, void* d_ws, size_t ws_size,
                              hipStream_t stream) {
    constexpr int N   = 16384;
    constexpr int E   = 524288;
    constexpr int DIN = 256;
    constexpr int H0  = 128;
    constexpr int H1  = 32;

    const float* feat      = (const float*)d_in[0];
    const int*   src       = (const int*)  d_in[1];
    const int*   dst       = (const int*)  d_in[2];
    const float* W0        = (const float*)d_in[3];
    const float* b0        = (const float*)d_in[4];
    const float* W1        = (const float*)d_in[5];
    const float* b1        = (const float*)d_in[6];
    const float* drop_mask = (const float*)d_in[7];
    float* adj             = (float*)d_out;

    // Workspace layout (aliased over time, ~25.2 MB total):
    char* ws = (char*)d_ws;
    float*  agg0_f32 = (float*) (ws);                       // 16 MB  [N*DIN f32]
    float*  h_f32    = (float*) (ws);                       //  8 MB  (reuses agg0 region)
    float*  agg1_f32 = (float*) (ws + (size_t)8  * 1048576);//  8 MB  [N*H0 f32]
    __bf16* agg0_b   = (__bf16*)(ws + (size_t)16 * 1048576);//  8 MB  [N*DIN bf16]
    __bf16* agg1_b   = (__bf16*)(ws + (size_t)16 * 1048576);//  4 MB  (reuses agg0_b region)
    __bf16* z_b      = (__bf16*)(ws + (size_t)24 * 1048576);//  1 MB  [N*H1 bf16]
    __bf16* W0t      = (__bf16*)(ws + (size_t)25 * 1048576);// 64 KB  [H0 x DIN bf16]
    __bf16* W1t      = (__bf16*)(ws + (size_t)25 * 1048576 + 65536); // 8 KB [H1 x H0]
    (void)in_sizes; (void)n_in; (void)out_size; (void)ws_size;

    // Layer 0 aggregation: agg0[dst] += feat[src]
    zero_f32<<<4096, 256, 0, stream>>>(agg0_f32, (size_t)N * DIN);
    scatter_add_edges<<<E, 256, 0, stream>>>(feat, src, dst, agg0_f32, DIN);
    cvt_f32_to_bf16<<<4096, 256, 0, stream>>>(agg0_f32, agg0_b, (size_t)N * DIN);

    // Weight prep (transpose -> bf16, contiguous-K for B fragments)
    transpose_to_bf16<<<(DIN * H0 + 255) / 256, 256, 0, stream>>>(W0, W0t, DIN, H0);
    transpose_to_bf16<<<(H0 * H1 + 255) / 256, 256, 0, stream>>>(W1, W1t, H0, H1);

    // h = relu(agg0 @ W0 + b0): M=16384, N=128, K=256
    //   tiles = (16384/16) * (128/64) = 2048 waves -> 256 blocks of 8 waves
    wmma_gemm_bt<4, true, DIN, false><<<256, 256, 0, stream>>>(
        agg0_b, W0t, b0, nullptr, h_f32, nullptr, N, H0);

    // Layer 1 aggregation: agg1[dst] += h[src]
    zero_f32<<<2048, 256, 0, stream>>>(agg1_f32, (size_t)N * H0);
    scatter_add_edges<<<E, 128, 0, stream>>>(h_f32, src, dst, agg1_f32, H0);
    cvt_f32_to_bf16<<<2048, 256, 0, stream>>>(agg1_f32, agg1_b, (size_t)N * H0);

    // z = (agg1 @ W1 + b1) * drop_mask: M=16384, N=32, K=128 -> bf16 output
    //   tiles = 1024 * (32/32) = 1024 waves -> 128 blocks
    wmma_gemm_bt<2, false, H0, false><<<128, 256, 0, stream>>>(
        agg1_b, W1t, b1, drop_mask, nullptr, z_b, N, H1);

    // adj = z @ z.T: M=N=16384, K=32 (one WMMA per 16x16 tile, fully unrolled)
    //   tiles = 1024 * (16384/64) = 262144 waves -> 32768 blocks
    //   non-temporal stores: stream the 1GB output without evicting z from L2
    wmma_gemm_bt<4, false, H1, true><<<32768, 256, 0, stream>>>(
        z_b, z_b, nullptr, nullptr, adj, nullptr, N, N);
}
